// DependencyTreeLSTM_26491358282205
// MI455X (gfx1250) — matrix operand order repaired and checked by reading, not compiled, not run on hardware
//
#include <hip/hip_runtime.h>
#include <hip/hip_bf16.h>

typedef __attribute__((ext_vector_type(16))) __bf16 v16bf;
typedef __attribute__((ext_vector_type(8)))  float  v8f;

#define LVL   12
#define NN    4096
#define KC    8
#define HD    300          // H == E == 300
#define KP    320          // padded K dimension (mult of 32)
#define HP    304          // padded output cols (mult of 16)
#define KT_N  10           // K tiles of 32
#define HT_N  19           // col tiles of 16
#define MATSZ ((size_t)HP * KP)

struct P8 { const float* p[8]; };
struct P4 { const float* p[4]; };

__device__ __forceinline__ float sigmoidf_(float x) { return 1.0f / (1.0f + __expf(-x)); }

// Build one lane's A-fragment (16x32 bf16 WMMA layout) from a row-major f32 row.
// Lane layout (this code's contiguous k order, matched in B): lanes 0-15 = M=0..15 K-half 0, lanes 16-31 = same M, K-half 1.
// elements 0..15 -> K = k0 + 16*half + t
__device__ __forceinline__ v16bf load_a_row_f32(const float* __restrict__ row, int k0, int half, bool tail) {
  v16bf a;
  int base = k0 + 16 * half;
  if (!tail) {
    const float4* p0 = (const float4*)(row + base);
    float4 r0 = p0[0], r1 = p0[1];
    const float4* p1 = (const float4*)(row + base + 8);
    float4 r2 = p1[0], r3 = p1[1];
    a[0] = (__bf16)r0.x;  a[1] = (__bf16)r0.y;  a[2]  = (__bf16)r0.z;  a[3]  = (__bf16)r0.w;
    a[4] = (__bf16)r1.x;  a[5] = (__bf16)r1.y;  a[6]  = (__bf16)r1.z;  a[7]  = (__bf16)r1.w;
    a[8] = (__bf16)r2.x;  a[9] = (__bf16)r2.y;  a[10] = (__bf16)r2.z;  a[11] = (__bf16)r2.w;
    a[12] = (__bf16)r3.x; a[13] = (__bf16)r3.y; a[14] = (__bf16)r3.z;  a[15] = (__bf16)r3.w;
  } else {
#pragma unroll
    for (int t = 0; t < 8; ++t) { int k = base + t;      a[t]     = (__bf16)((k < HD) ? row[k] : 0.0f); }
#pragma unroll
    for (int t = 0; t < 8; ++t) { int k = base + 8 + t; a[8 + t] = (__bf16)((k < HD) ? row[k] : 0.0f); }
  }
  return a;
}

// B-fragment (32x16 bf16): col n = lane&15, elements t -> K = k0 + 16*half + t.
// Weights stored row-major [HP][KP] bf16 (i.e. W itself, since GEMM is A @ W^T):
// a fragment is 16 consecutive bf16 = one aligned 32-byte load.
__device__ __forceinline__ v16bf load_b_w(const __bf16* __restrict__ wmat, int col, int k0, int half) {
  return *(const v16bf*)(wmat + (size_t)col * KP + k0 + 16 * half);
}

// ---- once per call: convert 8 weight matrices (300x300 f32) to padded bf16 [304][320]
__global__ void k_convw(P8 srcs, __bf16* __restrict__ out) {
  size_t total = 8 * MATSZ;
  for (size_t t = (size_t)blockIdx.x * blockDim.x + threadIdx.x; t < total;
       t += (size_t)gridDim.x * blockDim.x) {
    size_t m = t / MATSZ; size_t rem = t - m * MATSZ;
    int r = (int)(rem / KP); int k = (int)(rem - (size_t)r * KP);
    float v = (r < HD && k < HD) ? srcs.p[m][r * HD + k] : 0.0f;
    out[t] = (__bf16)v;
  }
}

__global__ void k_zero(float* __restrict__ a, int n) {
  for (int t = blockIdx.x * blockDim.x + threadIdx.x; t < n; t += gridDim.x * blockDim.x) a[t] = 0.0f;
}

// h_tilde[n] = sum_k mask[n][k] * h_prev[cidx[n][k]]
__global__ void k_htilde(const float* __restrict__ h_prev, const int* __restrict__ cidx,
                         const float* __restrict__ cmask, float* __restrict__ h_tilde) {
  int n = blockIdx.x;
  int idx[KC]; float mk[KC];
#pragma unroll
  for (int k = 0; k < KC; ++k) { idx[k] = cidx[n * KC + k] * HD; mk[k] = cmask[n * KC + k]; }
  for (int d = threadIdx.x; d < HD; d += blockDim.x) {
    float acc = 0.0f;
#pragma unroll
    for (int k = 0; k < KC; ++k) acc += mk[k] * h_prev[idx[k] + d];
    h_tilde[n * HD + d] = acc;
  }
}

// Shared-memory A-tile in fragment-linear order: slot (kt, lane) holds that lane's
// 16-element bf16 fragment for K-tile kt. 10 * 32 * 32B = 10 KB LDS.
#define SMA_ELEMS (KT_N * 32 * 16)

// wx[g] = embedding[wids] @ W_g^T + b_wg. One block per 16-row tile; A staged in LDS once,
// 8 waves sweep the 19 column tiles (4 gates per wave, shared A fragment).
__global__ void __launch_bounds__(256) k_gemm_wx(const float* __restrict__ emb,
                                                 const int* __restrict__ wids,
                                                 const __bf16* __restrict__ wts, P4 bw,
                                                 float* __restrict__ wx) {
  __shared__ __bf16 smA[SMA_ELEMS];
  int it = blockIdx.x;
  int lane = threadIdx.x;
  int tid = threadIdx.y * 32 + lane;
  for (int slot = tid; slot < KT_N * 32; slot += 256) {
    int kt = slot >> 5, sl = slot & 31;
    const float* row = emb + (size_t)wids[it * 16 + (sl & 15)] * HD;
    *(v16bf*)(smA + slot * 16) = load_a_row_f32(row, kt * 32, sl >> 4, kt == KT_N - 1);
  }
  __syncthreads();
  int half = lane >> 4, lm = lane & 15;
  v8f z = {0.f, 0.f, 0.f, 0.f, 0.f, 0.f, 0.f, 0.f};
  for (int jt = threadIdx.y; jt < HT_N; jt += 8) {
    int col = jt * 16 + lm;
    v8f acc[4] = {z, z, z, z};
    for (int kt = 0; kt < KT_N; ++kt) {
      v16bf a = *(const v16bf*)(smA + (kt * 32 + lane) * 16);
#pragma unroll
      for (int g = 0; g < 4; ++g) {
        v16bf b = load_b_w(wts + (size_t)g * MATSZ, col, kt * 32, half);
        acc[g] = __builtin_amdgcn_wmma_f32_16x16x32_bf16(false, a, false, b, (short)0, acc[g], false, false);
      }
    }
#pragma unroll
    for (int g = 0; g < 4; ++g) {
      float bias = (col < HD) ? bw.p[g][col] : 0.0f;
      float* outg = wx + (size_t)g * NN * HP;
#pragma unroll
      for (int v = 0; v < 8; ++v) {
        int rown = it * 16 + 8 * half + v;              // C layout: M = v + 8*half, N = lane&15
        outg[(size_t)rown * HP + col] = (col < HD) ? (acc[g][v] + bias) : 0.0f;
      }
    }
  }
}

// Per-child forget-gate GEMM (32768 gathered rows) fused with masked child-cell reduction:
// sum_c[node] = sum_k m * sigmoid(wx_f[node] + U_f h_k + b_uf) * c_prev[cidx].
// One block per 16-row tile (= 2 nodes x 8 children); A staged in LDS once.
__global__ void __launch_bounds__(256) k_gemm_f(const float* __restrict__ h_prev,
                                                const float* __restrict__ c_prev,
                                                const int* __restrict__ cidx,
                                                const float* __restrict__ cmask,
                                                const __bf16* __restrict__ uf,
                                                const float* __restrict__ b_uf,
                                                const float* __restrict__ wxf,
                                                float* __restrict__ sum_c) {
  __shared__ __bf16 smA[SMA_ELEMS];
  int it = blockIdx.x;
  int lane = threadIdx.x;
  int tid = threadIdx.y * 32 + lane;
  for (int slot = tid; slot < KT_N * 32; slot += 256) {
    int kt = slot >> 5, sl = slot & 31;
    const float* row = h_prev + (size_t)cidx[it * 16 + (sl & 15)] * HD;
    *(v16bf*)(smA + slot * 16) = load_a_row_f32(row, kt * 32, sl >> 4, kt == KT_N - 1);
  }
  __syncthreads();
  int half = lane >> 4, lm = lane & 15;
  // lane's 8 C rows = the 8 children of one node (rows ordered node*8+child): hoist gathers
  int node = it * 2 + half;
  int ci[8]; float mk[8];
#pragma unroll
  for (int v = 0; v < 8; ++v) {
    int grow = it * 16 + 8 * half + v;
    ci[v] = cidx[grow];
    mk[v] = cmask[grow];
  }
  v8f z = {0.f, 0.f, 0.f, 0.f, 0.f, 0.f, 0.f, 0.f};
  for (int jt = threadIdx.y; jt < HT_N; jt += 8) {
    int col = jt * 16 + lm;
    v8f acc = z;
    for (int kt = 0; kt < KT_N; ++kt) {
      v16bf a = *(const v16bf*)(smA + (kt * 32 + lane) * 16);
      v16bf b = load_b_w(uf, col, kt * 32, half);
      acc = __builtin_amdgcn_wmma_f32_16x16x32_bf16(false, a, false, b, (short)0, acc, false, false);
    }
    if (col < HD) {
      float wf = wxf[(size_t)node * HP + col];
      float bb = b_uf[col];
      float s = 0.0f;
#pragma unroll
      for (int v = 0; v < 8; ++v) {
        float cv = c_prev[(size_t)ci[v] * HD + col];
        s += mk[v] * sigmoidf_(acc[v] + wf + bb) * cv;
      }
      sum_c[(size_t)node * HD + col] = s;
    }
  }
}

// uh = h_tilde @ U_{i,o,u}^T fused with the LSTM pointwise epilogue -> h_next, c_next.
__global__ void __launch_bounds__(256) k_gemm_final(const float* __restrict__ h_tilde,
                                                    const __bf16* __restrict__ wts,
                                                    const float* __restrict__ b_ui,
                                                    const float* __restrict__ b_uo,
                                                    const float* __restrict__ b_uu,
                                                    const float* __restrict__ wx,
                                                    const float* __restrict__ sum_c,
                                                    float* __restrict__ h_next,
                                                    float* __restrict__ c_next) {
  __shared__ __bf16 smA[SMA_ELEMS];
  int it = blockIdx.x;
  int lane = threadIdx.x;
  int tid = threadIdx.y * 32 + lane;
  for (int slot = tid; slot < KT_N * 32; slot += 256) {
    int kt = slot >> 5, sl = slot & 31;
    const float* row = h_tilde + (size_t)(it * 16 + (sl & 15)) * HD;
    *(v16bf*)(smA + slot * 16) = load_a_row_f32(row, kt * 32, sl >> 4, kt == KT_N - 1);
  }
  __syncthreads();
  int half = lane >> 4, lm = lane & 15;
  const __bf16* ui = wts + (size_t)(4 + 0) * MATSZ;
  const __bf16* uo = wts + (size_t)(4 + 2) * MATSZ;
  const __bf16* uu = wts + (size_t)(4 + 3) * MATSZ;
  v8f z = {0.f, 0.f, 0.f, 0.f, 0.f, 0.f, 0.f, 0.f};
  for (int jt = threadIdx.y; jt < HT_N; jt += 8) {
    int col = jt * 16 + lm;
    v8f ai = z, ao = z, au = z;
    for (int kt = 0; kt < KT_N; ++kt) {
      v16bf a = *(const v16bf*)(smA + (kt * 32 + lane) * 16);
      v16bf bi = load_b_w(ui, col, kt * 32, half);
      ai = __builtin_amdgcn_wmma_f32_16x16x32_bf16(false, a, false, bi, (short)0, ai, false, false);
      v16bf bo = load_b_w(uo, col, kt * 32, half);
      ao = __builtin_amdgcn_wmma_f32_16x16x32_bf16(false, a, false, bo, (short)0, ao, false, false);
      v16bf bu = load_b_w(uu, col, kt * 32, half);
      au = __builtin_amdgcn_wmma_f32_16x16x32_bf16(false, a, false, bu, (short)0, au, false, false);
    }
    if (col < HD) {
      float bi_ = b_ui[col], bo_ = b_uo[col], bu_ = b_uu[col];
#pragma unroll
      for (int v = 0; v < 8; ++v) {
        int rown = it * 16 + 8 * half + v;
        float iv = sigmoidf_(wx[(size_t)0 * NN * HP + (size_t)rown * HP + col] + ai[v] + bi_);
        float ov = sigmoidf_(wx[(size_t)2 * NN * HP + (size_t)rown * HP + col] + ao[v] + bo_);
        float uv = tanhf   (wx[(size_t)3 * NN * HP + (size_t)rown * HP + col] + au[v] + bu_);
        float c = iv * uv + sum_c[(size_t)rown * HD + col];
        float h = ov * tanhf(c);
        c_next[(size_t)rown * HD + col] = c;
        h_next[(size_t)rown * HD + col] = h;
      }
    }
  }
}

extern "C" void kernel_launch(void* const* d_in, const int* in_sizes, int n_in,
                              void* d_out, int out_size, void* d_ws, size_t ws_size,
                              hipStream_t stream) {
  (void)in_sizes; (void)n_in; (void)out_size; (void)ws_size;
  const int*   word_ids   = (const int*)d_in[0];
  const int*   child_idx  = (const int*)d_in[1];
  const float* child_mask = (const float*)d_in[2];
  const float* emb        = (const float*)d_in[3];
  P8 wsrc; P4 bw; const float* bu[4];
  for (int g = 0; g < 4; ++g) {
    wsrc.p[g]     = (const float*)d_in[4 + 4 * g];   // w_g  [H,E]
    bw.p[g]       = (const float*)d_in[5 + 4 * g];   // b_wg [H]
    wsrc.p[4 + g] = (const float*)d_in[6 + 4 * g];   // u_g  [H,H]
    bu[g]         = (const float*)d_in[7 + 4 * g];   // b_ug [H]
  }

  char* base = (char*)d_ws;
  size_t off = 0;
  auto alloc = [&](size_t bytes) -> char* { char* p = base + off; off += (bytes + 255) & ~(size_t)255; return p; };
  __bf16* wts    = (__bf16*)alloc(8 * MATSZ * sizeof(__bf16));
  float* hbufA   = (float*)alloc((size_t)NN * HD * 4);
  float* hbufB   = (float*)alloc((size_t)NN * HD * 4);
  float* cbufA   = (float*)alloc((size_t)NN * HD * 4);
  float* cbufB   = (float*)alloc((size_t)NN * HD * 4);
  float* h_tilde = (float*)alloc((size_t)NN * HD * 4);
  float* wx      = (float*)alloc((size_t)4 * NN * HP * 4);
  float* sum_c   = (float*)alloc((size_t)NN * HD * 4);
  float* hbuf[2] = {hbufA, hbufB};
  float* cbuf[2] = {cbufA, cbufB};

  k_convw<<<512, 256, 0, stream>>>(wsrc, wts);
  k_zero<<<512, 256, 0, stream>>>(hbuf[0], NN * HD);
  k_zero<<<512, 256, 0, stream>>>(cbuf[0], NN * HD);

  dim3 blk(32, 8);
  for (int l = 0; l < LVL; ++l) {
    const int*   wids_l = word_ids + (size_t)l * NN;
    const int*   cidx_l = child_idx + (size_t)l * NN * KC;
    const float* cm_l   = child_mask + (size_t)l * NN * KC;
    float* hp = hbuf[l & 1];
    float* cp = cbuf[l & 1];
    float* hn = (l == LVL - 1) ? (float*)d_out : hbuf[(l + 1) & 1];
    float* cn = (l == LVL - 1) ? ((float*)d_out + (size_t)NN * HD) : cbuf[(l + 1) & 1];
    k_gemm_wx   <<<NN / 16, blk, 0, stream>>>(emb, wids_l, wts, bw, wx);
    k_htilde    <<<NN,      64,  0, stream>>>(hp, cidx_l, cm_l, h_tilde);
    k_gemm_f    <<<(NN * KC) / 16, blk, 0, stream>>>(hp, cp, cidx_l, cm_l,
                                                     wts + (size_t)5 * MATSZ, bu[1],
                                                     wx + (size_t)1 * NN * HP, sum_c);
    k_gemm_final<<<NN / 16, blk, 0, stream>>>(h_tilde, wts, bu[0], bu[2], bu[3], wx, sum_c, hn, cn);
  }
}
